// TextEncoder_13529146982818
// MI455X (gfx1250) — compile-verified, matrix operand order checked by reading, not run on hardware
//
#include <hip/hip_runtime.h>
#include <hip/hip_bf16.h>
#include <math.h>

// ---------------- problem constants ----------------
#define B_     8
#define T_     512
#define H_     512
#define HEADS_ 8
#define DH_    64
#define FFN_   2048
#define L_     6
#define KT_    3
#define OUT_   512
#define WIN_   4
#define SZH    (B_*H_*T_)          // 2,097,152 floats (8 MB)

typedef __attribute__((ext_vector_type(16))) _Float16 v16h;
typedef __attribute__((ext_vector_type(8)))  float    v8f;

// ---------------------------------------------------------------------------
// CDNA5 async global->LDS copy (ASYNCcnt-tracked). GVS addressing:
//   lds[VDST + 0] = mem[SADDR + VADDR]
// LDS address = low 32 bits of the generic pointer (flat->LDS truncation).
// ---------------------------------------------------------------------------
__device__ __forceinline__ void async_ld_b32(unsigned lds_off,
                                             unsigned long long sbase,
                                             int voff) {
  asm volatile("global_load_async_to_lds_b32 %0, %1, %2"
               :
               : "v"(lds_off), "v"(voff), "s"(sbase)
               : "memory");
}
#define WAIT_ASYNC_24() asm volatile("s_wait_asynccnt 0x18" ::: "memory")
#define WAIT_ASYNC_0()  asm volatile("s_wait_asynccnt 0x0" ::: "memory")

// ---------------------------------------------------------------------------
// Fragment loaders matching CDNA5 16-bit A/B WMMA VGPR layout (ISA 7.12.2):
// lane L holds row (L&15); K split in two 8-wide chunks at ksel=(L>>4)*8.
// Sources are f32 tiles in LDS (async-copied raw), converted to f16 here.
// ---------------------------------------------------------------------------
__device__ __forceinline__ v16h frag_f32(const float* rowp, int lane) {
  int ks = (lane >> 4) << 3;
  v16h r;
#pragma unroll
  for (int i = 0; i < 8; ++i) {
    r[i]     = (_Float16)rowp[ks + i];
    r[8 + i] = (_Float16)rowp[16 + ks + i];
  }
  return r;
}

__device__ __forceinline__ v16h frag_bf32(const float* rowp, int lane,
                                          bool valid) {
  int ks = (lane >> 4) << 3;
  v16h r;
#pragma unroll
  for (int i = 0; i < 8; ++i) {
    r[i]     = valid ? (_Float16)rowp[ks + i]      : (_Float16)0.0f;
    r[8 + i] = valid ? (_Float16)rowp[16 + ks + i] : (_Float16)0.0f;
  }
  return r;
}

// f16-source loader (attention keeps f16 staging)
__device__ __forceinline__ v16h frag_h(const _Float16* base, int row, int ld,
                                       int k0, int lane) {
  const _Float16* p = base + (size_t)row * ld + k0;
  int ks = (lane >> 4) << 3;
  v16h r;
#pragma unroll
  for (int i = 0; i < 8; ++i) { r[i] = p[ks + i]; r[8 + i] = p[16 + ks + i]; }
  return r;
}

__device__ __forceinline__ v16h frag_f(const float* rowp, int k0, int lane) {
  int ks = (lane >> 4) << 3;
  v16h r;
#pragma unroll
  for (int i = 0; i < 8; ++i) {
    r[i]     = (_Float16)rowp[k0 + ks + i];
    r[8 + i] = (_Float16)rowp[k0 + 16 + ks + i];
  }
  return r;
}

// ---------------------------------------------------------------------------
// Generic WMMA GEMM / 1D-conv kernel, double-buffered async-LDS pipeline.
//   Y[b,o,t] = bias[o] + sum_{tap,c} W[o,c,tap] * X[b,c,t+tap-(taps-1)/2]
// Input columns beyond lens[b] are zero (== h*mask), applied in the B loader.
// ---------------------------------------------------------------------------
#define TM 64
#define TN 128
#define WK 32

__global__ __launch_bounds__(256) void gemm_conv_wmma(
    const float* __restrict__ Wt, const float* __restrict__ bias,
    const float* __restrict__ X, float* __restrict__ Y,
    const int* __restrict__ lens, int O, int C, int taps, int relu,
    int maskout) {
  __shared__ float As[2][TM][WK];   // raw f32 weight tiles   (16 KB)
  __shared__ float Bs[2][TN][WK];   // raw f32 act tiles, [n][k] (32 KB)

  const int b    = blockIdx.z;
  const int m0   = blockIdx.y * TM;
  const int n0   = blockIdx.x * TN;
  const int tid  = threadIdx.x;
  const int lane = tid & 31;
  const int wave = tid >> 5;
  const int wm   = (wave >> 2) * 32;  // 0 | 32
  const int wn   = (wave & 3) * 32;   // 0..96
  const int len  = lens ? lens[b] : T_;
  const int half = (taps - 1) >> 1;

  const unsigned long long wbase = (unsigned long long)(size_t)Wt;
  const unsigned long long xbase =
      (unsigned long long)(size_t)(X + (size_t)b * C * T_);
  float* Yb = Y + (size_t)b * O * T_;

  const int kSteps = C / WK;
  const int steps  = taps * kSteps;

  // issue the 24 async copy instructions (per wave) for pipeline step s
  auto issue = [&](int s, int buf) {
    int tap    = s / kSteps;
    int c0     = (s - tap * kSteps) * WK;
    int tshift = tap - half;
#pragma unroll
    for (int r = 0; r < (TM * WK) / 256; ++r) {        // A: 8 instr/wave
      int idx = tid + r * 256;
      int mm = idx >> 5, kk = idx & 31;
      unsigned lo = (unsigned)(size_t)&As[buf][mm][kk];
      int go = (((m0 + mm) * C + (c0 + kk)) * taps + tap) << 2;
      async_ld_b32(lo, wbase, go);
    }
#pragma unroll
    for (int r = 0; r < (TN * WK) / 256; ++r) {        // B: 16 instr/wave
      int idx = tid + r * 256;
      int nn = idx >> 5, kk = idx & 31;
      int t = n0 + nn + tshift;
      t = t < 0 ? 0 : (t >= T_ ? T_ - 1 : t);          // clamp; mask in loader
      unsigned lo = (unsigned)(size_t)&Bs[buf][nn][kk];
      int go = ((c0 + kk) * T_ + t) << 2;
      async_ld_b32(lo, xbase, go);
    }
  };

  v8f zero = {};
  v8f acc[2][2];
#pragma unroll
  for (int i = 0; i < 2; ++i)
#pragma unroll
    for (int j = 0; j < 2; ++j) acc[i][j] = zero;

  issue(0, 0);
  for (int s = 0; s < steps; ++s) {
    const int buf    = s & 1;
    const int tap    = s / kSteps;
    const int tshift = tap - half;
    if (s + 1 < steps) {
      issue(s + 1, buf ^ 1);
      WAIT_ASYNC_24();   // own 24 new ops may be pending; tile s is complete
    } else {
      WAIT_ASYNC_0();
    }
    __syncthreads();

    v16h bfr[2];
#pragma unroll
    for (int j = 0; j < 2; ++j) {
      int nn = wn + j * 16 + (lane & 15);
      int tcol = n0 + nn + tshift;
      bool valid = (tcol >= 0) && (tcol < T_) && (tcol < len);
      bfr[j] = frag_bf32(&Bs[buf][nn][0], lane, valid);
    }
#pragma unroll
    for (int i = 0; i < 2; ++i) {
      v16h a = frag_f32(&As[buf][wm + i * 16 + (lane & 15)][0], lane);
#pragma unroll
      for (int j = 0; j < 2; ++j) {
        acc[i][j] = __builtin_amdgcn_wmma_f32_16x16x32_f16(
            false, a, false, bfr[j], (short)0, acc[i][j], false, false);
      }
    }
    __syncthreads();   // protect buf before it is overwritten at step s+2
  }

  const int rbase = (lane < 16) ? 0 : 8;
#pragma unroll
  for (int i = 0; i < 2; ++i) {
#pragma unroll
    for (int j = 0; j < 2; ++j) {
      int ncol = n0 + wn + j * 16 + (lane & 15);
#pragma unroll
      for (int vv = 0; vv < 8; ++vv) {
        int mr = m0 + wm + i * 16 + rbase + vv;
        float val = acc[i][j][vv] + (bias ? bias[mr] : 0.0f);
        if (relu) val = fmaxf(val, 0.0f);
        if (maskout && ncol >= len) val = 0.0f;
        Yb[(size_t)mr * T_ + ncol] = val;
      }
    }
  }
}

// ---------------------------------------------------------------------------
// Fused relative-position attention for one (b, head, 16-query tile).
// rel_to_abs / abs_to_rel collapse to a +-WIN band:
//   scores[t,s] += q[t]·rel_k[s-t+W]   for |s-t|<=W
//   o[t,d]      += p[t,s]·rel_v[s-t+W] for |s-t|<=W
// ---------------------------------------------------------------------------
__global__ __launch_bounds__(256) void attn_kernel(
    const float* __restrict__ qg, const float* __restrict__ kg,
    const float* __restrict__ vg, float* __restrict__ og,
    const float* __restrict__ relk, const float* __restrict__ relv,
    const int* __restrict__ lens) {
  __shared__ _Float16 qs[16][64];       // scaled Q tile (f16)
  __shared__ float    sc[16][T_];       // scores -> probabilities
  __shared__ _Float16 stage[128][32];   // K / V^T staging
  __shared__ float    ot[16][64];       // output tile before rel_v band

  const int t0   = blockIdx.x * 16;
  const int head = blockIdx.y;
  const int b    = blockIdx.z;
  const int tid  = threadIdx.x;
  const int lane = tid & 31;
  const int wave = tid >> 5;
  const int len  = lens[b];

  const size_t hb = ((size_t)b * H_ + head * DH_) * T_;
  const float* qb = qg + hb;
  const float* kb = kg + hb;
  const float* vb = vg + hb;
  float*       ob = og + hb;

  for (int r = tid; r < 16 * 64; r += 256) {
    int tl = r >> 6, d = r & 63;
    qs[tl][d] = (_Float16)(qb[(size_t)d * T_ + (t0 + tl)] * 0.125f);
  }
  __syncthreads();

  // scores = Q K^T : each wave owns one 16-col slice of a 16x128 stripe
  for (int s0 = 0; s0 < T_; s0 += 128) {
    v8f acc = {};
#pragma unroll
    for (int dhalf = 0; dhalf < 2; ++dhalf) {
      int d0 = dhalf * 32;
      __syncthreads();
      for (int r = tid; r < 128 * 32; r += 256) {
        int nn = r >> 5, kk = r & 31;
        stage[nn][kk] = (_Float16)kb[(size_t)(d0 + kk) * T_ + (s0 + nn)];
      }
      __syncthreads();
      v16h a  = frag_h(&qs[0][0], lane & 15, 64, d0, lane);
      v16h bb = frag_h(&stage[0][0], wave * 16 + (lane & 15), 32, 0, lane);
      acc = __builtin_amdgcn_wmma_f32_16x16x32_f16(false, a, false, bb,
                                                   (short)0, acc, false, false);
    }
    const int col   = s0 + wave * 16 + (lane & 15);
    const int rbase = (lane < 16) ? 0 : 8;
#pragma unroll
    for (int vv = 0; vv < 8; ++vv) sc[rbase + vv][col] = acc[vv];
  }
  __syncthreads();

  // rel_k band add
  for (int task = tid; task < 16 * (2 * WIN_ + 1); task += 256) {
    int tl = task / (2 * WIN_ + 1);
    int dd = task % (2 * WIN_ + 1);
    int s = t0 + tl + dd - WIN_;
    if (s >= 0 && s < T_) {
      float dot = 0.0f;
#pragma unroll
      for (int d = 0; d < DH_; ++d)
        dot += (float)qs[tl][d] * relk[dd * DH_ + d];
      sc[tl][s] += dot;
    }
  }
  __syncthreads();

  // mask + softmax : one wave per pair of rows
  for (int rr = 0; rr < 2; ++rr) {
    int tl = wave * 2 + rr;
    int tglob = t0 + tl;
    float mx = -1e30f;
    for (int s = lane; s < T_; s += 32) {
      float val = sc[tl][s];
      if (s >= len || tglob >= len) val = -10000.0f;
      sc[tl][s] = val;
      mx = fmaxf(mx, val);
    }
#pragma unroll
    for (int off = 16; off; off >>= 1) mx = fmaxf(mx, __shfl_xor(mx, off, 32));
    float sum = 0.0f;
    for (int s = lane; s < T_; s += 32) {
      float e = __expf(sc[tl][s] - mx);
      sc[tl][s] = e;
      sum += e;
    }
#pragma unroll
    for (int off = 16; off; off >>= 1) sum += __shfl_xor(sum, off, 32);
    float inv = 1.0f / sum;
    for (int s = lane; s < T_; s += 32) sc[tl][s] *= inv;
  }
  __syncthreads();

  // O = P V : 16x64 output, waves 0..3 each own one 16-col fragment
  v8f accO = {};
  for (int s0 = 0; s0 < T_; s0 += 32) {
    __syncthreads();
    for (int r = tid; r < 64 * 32; r += 256) {
      int d = r >> 5, ss = r & 31;
      stage[d][ss] = (_Float16)vb[(size_t)d * T_ + (s0 + ss)];
    }
    __syncthreads();
    if (wave < 4) {
      v16h a  = frag_f(&sc[lane & 15][0], s0, lane);
      v16h bb = frag_h(&stage[0][0], wave * 16 + (lane & 15), 32, 0, lane);
      accO = __builtin_amdgcn_wmma_f32_16x16x32_f16(
          false, a, false, bb, (short)0, accO, false, false);
    }
  }
  if (wave < 4) {
    const int col   = wave * 16 + (lane & 15);
    const int rbase = (lane < 16) ? 0 : 8;
#pragma unroll
    for (int vv = 0; vv < 8; ++vv) ot[rbase + vv][col] = accO[vv];
  }
  __syncthreads();

  // rel_v band add + store in (B, H, T) layout for the O-projection GEMM
  for (int task = tid; task < 16 * 64; task += 256) {
    int tl = task >> 6, d = task & 63;
    int tglob = t0 + tl;
    float val = ot[tl][d];
#pragma unroll
    for (int dd = 0; dd < 2 * WIN_ + 1; ++dd) {
      int s = tglob + dd - WIN_;
      if (s >= 0 && s < T_) val += sc[tl][s] * relv[dd * DH_ + d];
    }
    ob[(size_t)d * T_ + tglob] = val;
  }
}

// ---------------------------------------------------------------------------
// h = LayerNorm(h + y) over channel dim (H=512); one wave per (b,t) position.
// ---------------------------------------------------------------------------
__global__ __launch_bounds__(256) void ln_residual(
    float* __restrict__ h, const float* __restrict__ y,
    const float* __restrict__ g, const float* __restrict__ be) {
  const int wave = threadIdx.x >> 5, lane = threadIdx.x & 31;
  const int p = blockIdx.x * 8 + wave;  // p in [0, B*T)
  const int b = p / T_, t = p % T_;
  float* hb = h + ((size_t)b * H_) * T_ + t;
  const float* yb = y + ((size_t)b * H_) * T_ + t;

  float vals[16];
  float s = 0.0f;
#pragma unroll
  for (int i = 0; i < 16; ++i) {
    int c = lane + i * 32;
    float v0 = hb[(size_t)c * T_] + yb[(size_t)c * T_];
    vals[i] = v0;
    s += v0;
  }
#pragma unroll
  for (int off = 16; off; off >>= 1) s += __shfl_xor(s, off, 32);
  float mu = s * (1.0f / H_);
  float var = 0.0f;
#pragma unroll
  for (int i = 0; i < 16; ++i) {
    float d = vals[i] - mu;
    var += d * d;
  }
#pragma unroll
  for (int off = 16; off; off >>= 1) var += __shfl_xor(var, off, 32);
  var *= (1.0f / H_);
  float rs = rsqrtf(var + 1e-5f);
#pragma unroll
  for (int i = 0; i < 16; ++i) {
    int c = lane + i * 32;
    hb[(size_t)c * T_] = (vals[i] - mu) * rs * g[c] + be[c];
  }
}

// ---------------------------------------------------------------------------
__global__ __launch_bounds__(256) void embed_kernel(
    const int* __restrict__ x, const float* __restrict__ emb,
    float* __restrict__ h) {
  int idx = blockIdx.x * 256 + threadIdx.x;
  if (idx >= SZH) return;
  int t = idx % T_;
  int c = (idx / T_) % H_;
  int b = idx / (T_ * H_);
  int tok = x[b * T_ + t];
  h[idx] = emb[(size_t)tok * H_ + c] * 22.62741699796952f;  // sqrt(512)
}

__global__ __launch_bounds__(256) void finalize_kernel(
    const float* __restrict__ h, const float* __restrict__ stats,
    const int* __restrict__ lens, float* __restrict__ out) {
  int idx = blockIdx.x * 256 + threadIdx.x;
  if (idx >= SZH) return;
  int t = idx % T_;
  int c = (idx / T_) % H_;
  int b = idx / (T_ * H_);
  float mval = (t < lens[b]) ? 1.0f : 0.0f;
  size_t so = ((size_t)b * (2 * OUT_) + c) * T_ + t;
  out[idx]           = h[idx] * mval;                          // h * mask
  out[SZH + idx]     = stats[so] * mval;                       // m
  out[2 * SZH + idx] = stats[so + (size_t)OUT_ * T_] * mval;   // logs
  if (c == 0) out[3 * (size_t)SZH + (size_t)b * T_ + t] = mval;  // mask
}

// ---------------------------------------------------------------------------
extern "C" void kernel_launch(void* const* d_in, const int* in_sizes, int n_in,
                              void* d_out, int out_size, void* d_ws,
                              size_t ws_size, hipStream_t stream) {
  (void)in_sizes; (void)n_in; (void)out_size; (void)ws_size;
  const int*   x    = (const int*)d_in[0];
  const int*   xlen = (const int*)d_in[1];
  const float* emb  = (const float*)d_in[2];
  const float* wq   = (const float*)d_in[3];
  const float* bq   = (const float*)d_in[4];
  const float* wk   = (const float*)d_in[5];
  const float* bk   = (const float*)d_in[6];
  const float* wv   = (const float*)d_in[7];
  const float* bv   = (const float*)d_in[8];
  const float* wo   = (const float*)d_in[9];
  const float* bo   = (const float*)d_in[10];
  const float* relk = (const float*)d_in[11];
  const float* relv = (const float*)d_in[12];
  const float* g1   = (const float*)d_in[13];
  const float* be1  = (const float*)d_in[14];
  const float* w1   = (const float*)d_in[15];
  const float* fb1  = (const float*)d_in[16];
  const float* w2   = (const float*)d_in[17];
  const float* fb2  = (const float*)d_in[18];
  const float* g2   = (const float*)d_in[19];
  const float* be2  = (const float*)d_in[20];
  const float* pw   = (const float*)d_in[21];
  const float* pb   = (const float*)d_in[22];

  float* ws = (float*)d_ws;
  float* h = ws;
  float* y = h + SZH;
  float* q = y + SZH;
  float* k = q + SZH;
  float* v = k + SZH;
  float* o = v + SZH;
  float* f = o + SZH;          // B*FFN*T
  float* stats = q;            // reuse q+k (4,194,304 floats) after the loop

  dim3 blk(256);
  dim3 gProj(T_ / TN, H_ / TM, B_);
  dim3 gF1(T_ / TN, FFN_ / TM, B_);
  dim3 gF2(T_ / TN, H_ / TM, B_);
  dim3 gStats(T_ / TN, (2 * OUT_) / TM, B_);
  dim3 gAttn(T_ / 16, HEADS_, B_);
  dim3 gLn(B_ * T_ / 8);
  int eBlocks = SZH / 256;

  embed_kernel<<<eBlocks, blk, 0, stream>>>(x, emb, h);

  for (int i = 0; i < L_; ++i) {
    const float* wq_i = wq + (size_t)i * H_ * H_;
    const float* wk_i = wk + (size_t)i * H_ * H_;
    const float* wv_i = wv + (size_t)i * H_ * H_;
    const float* wo_i = wo + (size_t)i * H_ * H_;
    const float* bq_i = bq + (size_t)i * H_;
    const float* bk_i = bk + (size_t)i * H_;
    const float* bv_i = bv + (size_t)i * H_;
    const float* bo_i = bo + (size_t)i * H_;
    const float* rk_i = relk + (size_t)i * (2 * WIN_ + 1) * DH_;
    const float* rv_i = relv + (size_t)i * (2 * WIN_ + 1) * DH_;
    const float* w1_i = w1 + (size_t)i * FFN_ * H_ * KT_;
    const float* w2_i = w2 + (size_t)i * H_ * FFN_ * KT_;

    gemm_conv_wmma<<<gProj, blk, 0, stream>>>(wq_i, bq_i, h, q, xlen,
                                              H_, H_, 1, 0, 0);
    gemm_conv_wmma<<<gProj, blk, 0, stream>>>(wk_i, bk_i, h, k, xlen,
                                              H_, H_, 1, 0, 0);
    gemm_conv_wmma<<<gProj, blk, 0, stream>>>(wv_i, bv_i, h, v, xlen,
                                              H_, H_, 1, 0, 0);
    attn_kernel<<<gAttn, blk, 0, stream>>>(q, k, v, o, rk_i, rv_i, xlen);
    gemm_conv_wmma<<<gProj, blk, 0, stream>>>(wo_i, bo_i, o, y, nullptr,
                                              H_, H_, 1, 0, 0);
    ln_residual<<<gLn, blk, 0, stream>>>(h, y, g1 + (size_t)i * H_,
                                         be1 + (size_t)i * H_);
    gemm_conv_wmma<<<gF1, blk, 0, stream>>>(w1_i, fb1 + (size_t)i * FFN_, h, f,
                                            xlen, FFN_, H_, KT_, 1, 0);
    gemm_conv_wmma<<<gF2, blk, 0, stream>>>(w2_i, fb2 + (size_t)i * H_, f, y,
                                            xlen, H_, FFN_, KT_, 0, 1);
    ln_residual<<<gLn, blk, 0, stream>>>(h, y, g2 + (size_t)i * H_,
                                         be2 + (size_t)i * H_);
  }

  gemm_conv_wmma<<<gStats, blk, 0, stream>>>(pw, pb, h, stats, xlen,
                                             2 * OUT_, H_, 1, 0, 1);
  finalize_kernel<<<eBlocks, blk, 0, stream>>>(h, stats, xlen, (float*)d_out);
}
